// AdaptiveSparseAttention_24687472017507
// MI455X (gfx1250) — matrix operand order repaired and checked by reading, not compiled, not run on hardware
//
#include <hip/hip_runtime.h>
#include <cstdint>

#define S_LEN 2048
#define BATCH 2
#define DMODEL 1024
#define NHEADS 16
#define HEAD_DK 64
#define HI_N 4
#define DI_N 64
#define TOPK 512
#define ROWS (BATCH * S_LEN) /* 4096 */
#define KCHUNK 64            /* k-chunk staged in LDS per GEMM iteration */

typedef unsigned short u16;
typedef __attribute__((ext_vector_type(16))) __bf16 v16bf;
typedef __attribute__((ext_vector_type(8))) float v8f;
typedef __attribute__((ext_vector_type(4))) unsigned int v4u;
typedef __attribute__((ext_vector_type(8))) int v8i;
typedef __attribute__((ext_vector_type(4))) int v4i;

#if defined(__has_builtin)
#if __has_builtin(__builtin_amdgcn_tensor_load_to_lds) && __has_builtin(__builtin_amdgcn_s_wait_tensorcnt)
#define USE_TDM 1
#endif
#endif
#ifndef USE_TDM
#define USE_TDM 0
#endif

__device__ __forceinline__ float bf2f(unsigned v) { return __uint_as_float(v << 16); }
__device__ __forceinline__ u16 f2bf(float f) {
  unsigned b = __float_as_uint(f);
  b += 0x7fffu + ((b >> 16) & 1u);
  return (u16)(b >> 16);
}

// ---------------------------------------------------------------- conversion
__global__ void cvt_f32_bf16(const float* __restrict__ in, u16* __restrict__ out, int n) {
  int i = blockIdx.x * blockDim.x + threadIdx.x;
  if (i < n) out[i] = f2bf(in[i]);
}

// ------------------------------------------------------- WMMA fragment loads
// A (16x32 bf16): lanes 0-15 -> M=lane, K-base 0; lanes 16-31 -> K-base 16.
// elements 0..15 = K kb..kb+15  (two 16B loads)
__device__ __forceinline__ v16bf load_a_frag(const u16* base, int ld, int r0, int k0, int lane) {
  int r = r0 + (lane & 15);
  int kb = k0 + ((lane >> 4) << 4);
  const u16* p = base + (size_t)r * ld + kb;
  union { v16bf v; uint4 u[2]; } f;
  f.u[0] = *(const uint4*)p;
  f.u[1] = *(const uint4*)(p + 8);
  return f.v;
}
// B (32x16 bf16): lanes 0-15 -> N=lane, K=0..15; lanes 16-31 -> K=16..31 (contiguous 32B)
__device__ __forceinline__ v16bf load_b_frag(const u16* base, int ld, int n0, int k0, int lane) {
  int r = n0 + (lane & 15);
  int kb = k0 + ((lane >> 4) << 4);
  const u16* p = base + (size_t)r * ld + kb;
  union { v16bf v; uint4 u[2]; } f;
  f.u[0] = *(const uint4*)p;
  f.u[1] = *(const uint4*)(p + 8);
  return f.v;
}
// B fragment out of the LDS-staged 64x64 bf16 tile (row-major, ld = KCHUNK)
__device__ __forceinline__ v16bf load_b_frag_lds(const u16* s, int n0l, int k0, int lane) {
  int r = n0l + (lane & 15);
  int kb = k0 + ((lane >> 4) << 4);
  const u16* p = s + r * KCHUNK + kb;
  union { v16bf v; uint4 u[2]; } f;
  f.u[0] = *(const uint4*)p;
  f.u[1] = *(const uint4*)(p + 8);
  return f.v;
}

// -------------------------------------------------- Tensor Data Mover (TDM)
// 2D descriptor: data_size=8B, tile = 8 units x 64 rows (64B x 64, half of the 128B-wide B k-chunk),
// row stride = Kd*2/8 units.  Groups per CDNA5 ISA.
#if USE_TDM
__device__ __forceinline__ void tdm_load_b_tile(const u16* gtile, unsigned lds_off, int Kd) {
  unsigned long long ga = (unsigned long long)(uintptr_t)gtile;
  v4u g0;
  g0[0] = 1u;                                                  // count=1, user mode
  g0[1] = lds_off;                                             // lds_addr (bytes)
  g0[2] = (unsigned)(ga & 0xffffffffu);                        // global_addr[31:0]
  g0[3] = (unsigned)((ga >> 32) & 0x01ffffffu) | (2u << 30);   // addr[56:32] | type=2
  v8i g1;
  g1[0] = (int)(3u << 16);       // workgroup_mask=0, data_size=3 (8B)
  g1[1] = (int)(8u << 16);       // tensor_dim0 = 8 units
  g1[2] = (int)(64u << 16);      // tensor_dim1 = 64 rows
  g1[3] = (int)(8u << 16);       // tile_dim0 = 8 units
  g1[4] = 64;                    // tile_dim1 = 64 rows, tile_dim2 = 0
  g1[5] = (int)((unsigned)Kd >> 2); // tensor_dim0_stride = Kd*2/8 units
  g1[6] = 0;
  g1[7] = 0;
  v4i z4 = {0, 0, 0, 0};
#if __clang_major__ >= 23
  v8i z8 = {0, 0, 0, 0, 0, 0, 0, 0};
  __builtin_amdgcn_tensor_load_to_lds(g0, g1, z4, z4, z8, 0);
#else
  __builtin_amdgcn_tensor_load_to_lds(g0, g1, z4, z4, 0);
#endif
}
#endif

// --------------------------------------------------------------- WMMA GEMM
// C(M,N) = A(M,K) * B(N,K)^T ; block = 4 waves, 128x64 tile; B k-chunks staged in
// LDS (TDM, double buffered) so the next fill overlaps 16 back-to-back WMMAs.
// Requires M % 128 == 0 and Kd % KCHUNK == 0 (true for all uses here).
__global__ void gemm_bf16_wmma(const u16* __restrict__ A, const u16* __restrict__ B,
                               float* __restrict__ Cf, u16* __restrict__ Cb,
                               int M, int N, int Kd, int store_bf16) {
  __shared__ u16 bsm[2][64 * KCHUNK]; // 2 x 8KB
  int lane = threadIdx.x & 31;
  int wave = threadIdx.x >> 5;
  int m0 = blockIdx.y * 128 + wave * 32;
  int n0 = blockIdx.x * 64;
  const u16* Btile = B + (size_t)n0 * Kd;

  v8f acc[2][4];
#pragma unroll
  for (int i = 0; i < 2; ++i)
#pragma unroll
    for (int j = 0; j < 4; ++j) acc[i][j] = {};

  int nch = Kd / KCHUNK;
#if USE_TDM
  if (wave == 0) tdm_load_b_tile(Btile, (unsigned)(uintptr_t)&bsm[0][0], Kd);
#endif
  for (int c = 0; c < nch; ++c) {
    int buf = c & 1;
#if USE_TDM
    if (wave == 0) {
      if (c + 1 < nch) {
        tdm_load_b_tile(Btile + (c + 1) * KCHUNK, (unsigned)(uintptr_t)&bsm[buf ^ 1][0], Kd);
        __builtin_amdgcn_s_wait_tensorcnt(1); // chunk c landed; chunk c+1 in flight
      } else {
        __builtin_amdgcn_s_wait_tensorcnt(0); // last chunk landed
      }
    }
#else
    // fallback: cooperative copy of the 64x64 bf16 chunk
    for (int i = threadIdx.x; i < 64 * (KCHUNK / 8); i += 128) {
      int r = i >> 3;
      int kk = (i & 7) * 8;
      *(uint4*)&bsm[buf][r * KCHUNK + kk] =
          *(const uint4*)(Btile + (size_t)r * Kd + c * KCHUNK + kk);
    }
#endif
    __syncthreads();
    int kbase = c * KCHUNK;
    __builtin_prefetch(A + (size_t)m0 * Kd + kbase + 2 * KCHUNK, 0, 1); // global_prefetch_b8
    v16bf a[2][2];
    v16bf b[2][4];
#pragma unroll
    for (int ks = 0; ks < 2; ++ks) {
      a[ks][0] = load_a_frag(A, Kd, m0, kbase + ks * 32, lane);
      a[ks][1] = load_a_frag(A, Kd, m0 + 16, kbase + ks * 32, lane);
#pragma unroll
      for (int j = 0; j < 4; ++j) b[ks][j] = load_b_frag_lds(&bsm[buf][0], 16 * j, ks * 32, lane);
    }
#pragma unroll
    for (int ks = 0; ks < 2; ++ks)
#pragma unroll
      for (int j = 0; j < 4; ++j) {
        acc[0][j] = __builtin_amdgcn_wmma_f32_16x16x32_bf16(false, a[ks][0], false, b[ks][j], (short)0, acc[0][j], false, false);
        acc[1][j] = __builtin_amdgcn_wmma_f32_16x16x32_bf16(false, a[ks][1], false, b[ks][j], (short)0, acc[1][j], false, false);
      }
    __syncthreads();
  }

  int rofs = (lane >> 4) << 3;
  int col = lane & 15;
#pragma unroll
  for (int i = 0; i < 2; ++i)
#pragma unroll
    for (int j = 0; j < 4; ++j)
#pragma unroll
      for (int v = 0; v < 8; ++v) {
        int m = m0 + 16 * i + rofs + v;
        int n = n0 + 16 * j + col;
        float val = acc[i][j][v];
        if (store_bf16) Cb[(size_t)m * N + n] = f2bf(val);
        else            Cf[(size_t)m * N + n] = val;
      }
}

// ------------------------------------------------------------ RoPE + split
__global__ void rope_split(const float* __restrict__ qkv, u16* __restrict__ qb,
                           u16* __restrict__ kb, u16* __restrict__ vb) {
  int idx = blockIdx.x * blockDim.x + threadIdx.x;
  if (idx >= ROWS * NHEADS * 32) return;
  int i = idx & 31;
  int h = (idx >> 5) & 15;
  int row = idx >> 9;
  int s = row & (S_LEN - 1);
  float inv = __powf(10000.f, -((float)(2 * i)) / 64.f);
  float sn, cs;
  __sincosf((float)s * inv, &sn, &cs);
  const float* bq = qkv + (size_t)row * (3 * DMODEL) + h * 64 + 2 * i;
  float q1 = bq[0], q2 = bq[1];
  float k1 = bq[DMODEL], k2 = bq[DMODEL + 1];
  float v1 = bq[2 * DMODEL], v2 = bq[2 * DMODEL + 1];
  size_t o = (size_t)(row * NHEADS + h) * 64 + 2 * i;
  qb[o] = f2bf(q1 * cs - q2 * sn);
  qb[o + 1] = f2bf(q1 * sn + q2 * cs);
  kb[o] = f2bf(k1 * cs - k2 * sn);
  kb[o + 1] = f2bf(k1 * sn + k2 * cs);
  vb[o] = f2bf(v1);
  vb[o + 1] = f2bf(v2);
}

// ---------------------------------------------------------------- wi = x@w_wi^T
__global__ void wi_proj(const float* __restrict__ x, const float* __restrict__ w,
                        float* __restrict__ wi) {
  int idx = blockIdx.x * blockDim.x + threadIdx.x;
  if (idx >= ROWS * HI_N) return;
  int h = idx & 3;
  int row = idx >> 2;
  const float4* xr = (const float4*)(x + (size_t)row * DMODEL);
  const float4* wr = (const float4*)(w + (size_t)h * DMODEL);
  float acc = 0.f;
  for (int d = 0; d < DMODEL / 4; ++d) {
    float4 a = xr[d], b = wr[d];
    acc += a.x * b.x + a.y * b.y + a.z * b.z + a.w * b.w;
  }
  wi[idx] = acc;
}

// ------------------------------------------------- indexer scores (causal tiles)
__global__ void indexer_scores(const u16* __restrict__ qi, const u16* __restrict__ ki,
                               const float* __restrict__ wi, float* __restrict__ sc) {
  int st = blockIdx.x, tt = blockIdx.y, b = blockIdx.z;
  if (st > tt) return;
  int lane = threadIdx.x;
  int t0 = tt * 16, s0 = st * 16;
  const u16* Bb = ki + (size_t)b * S_LEN * DI_N;
  v16bf b0 = load_b_frag(Bb, DI_N, s0, 0, lane);
  v16bf b1 = load_b_frag(Bb, DI_N, s0, 32, lane);
  float out[8];
#pragma unroll
  for (int v = 0; v < 8; ++v) out[v] = 0.f;
  int rofs = (lane >> 4) << 3;
  int col = lane & 15;
#pragma unroll
  for (int h = 0; h < HI_N; ++h) {
    const u16* Ab = qi + (size_t)b * S_LEN * (HI_N * DI_N) + h * DI_N;
    v16bf a0 = load_a_frag(Ab, HI_N * DI_N, t0, 0, lane);
    v16bf a1 = load_a_frag(Ab, HI_N * DI_N, t0, 32, lane);
    v8f c = {};
    c = __builtin_amdgcn_wmma_f32_16x16x32_bf16(false, a0, false, b0, (short)0, c, false, false);
    c = __builtin_amdgcn_wmma_f32_16x16x32_bf16(false, a1, false, b1, (short)0, c, false, false);
#pragma unroll
    for (int v = 0; v < 8; ++v) {
      float wv = wi[((size_t)b * S_LEN + t0 + rofs + v) * HI_N + h];
      float r = c[v] > 0.f ? c[v] : 0.f;
      out[v] += r * wv;
    }
  }
#pragma unroll
  for (int v = 0; v < 8; ++v)
    sc[((size_t)b * S_LEN + t0 + rofs + v) * S_LEN + s0 + col] = out[v];
}

// ------------------------------------------------- per-row radix top-K select
__global__ void topk_select(const float* __restrict__ sc, int* __restrict__ idxl) {
  __shared__ unsigned u[S_LEN];
  __shared__ unsigned hist[256];
  int lane = threadIdx.x;
  int row = blockIdx.x;         // b*S + t
  int t = row & (S_LEN - 1);
  if (t < TOPK) {               // top_k of masked row picks {0..K-1} (stable ties)
    for (int j = lane; j < TOPK; j += 32) idxl[(size_t)row * TOPK + j] = j;
    return;
  }
  int n = t + 1;
  const float* s = sc + (size_t)row * S_LEN;
  for (int c = lane; c < n; c += 32) {
    unsigned bits = __float_as_uint(s[c]);
    u[c] = (bits & 0x80000000u) ? ~bits : (bits | 0x80000000u);
  }
  __syncthreads();
  unsigned prefix = 0, pmask = 0;
  int need = TOPK;
  for (int level = 3; level >= 0; --level) {
    int shift = level * 8;
    for (int d = lane; d < 256; d += 32) hist[d] = 0u;
    __syncthreads();
    for (int c = lane; c < n; c += 32) {
      unsigned uv = u[c];
      if ((uv & pmask) == prefix) atomicAdd(&hist[(uv >> shift) & 255], 1u);
    }
    __syncthreads();
    int cum = 0, chosen = 0, before = 0;   // uniform scan (same on all lanes)
    for (int d = 255; d >= 0; --d) {
      int hv = (int)hist[d];
      if (cum + hv >= need) { chosen = d; before = cum; break; }
      cum += hv;
    }
    need -= before;
    prefix |= ((unsigned)chosen) << shift;
    pmask |= 255u << shift;
    __syncthreads();
  }
  unsigned thr = prefix;
  int needEq = need;
  int selBase = 0, eqBase = 0;
  for (int c0 = 0; c0 < n; c0 += 32) {
    int c = c0 + lane;
    bool act = c < n;
    unsigned uv = act ? u[c] : 0u;
    bool g = act && (uv > thr);
    bool e = act && (uv == thr);
    unsigned em = (unsigned)__ballot(e);
    int eqrank = __popc(em & ((1u << lane) - 1u));
    bool sel = g || (e && (eqBase + eqrank) < needEq);
    unsigned sm = (unsigned)__ballot(sel);
    int sr = __popc(sm & ((1u << lane) - 1u));
    if (sel) idxl[(size_t)row * TOPK + selBase + sr] = c;
    selBase += __popc(sm);
    eqBase += __popc(em);
  }
}

// ------------------------------------------------- sparse gather attention
// block = 512 threads = 16 waves, one wave per head; one block per (b,t)
__global__ void attention_sparse(const u16* __restrict__ qb, const u16* __restrict__ kb,
                                 const u16* __restrict__ vb, const int* __restrict__ idxl,
                                 u16* __restrict__ ob) {
  __shared__ float qs[NHEADS * 64];
  __shared__ float ps[NHEADS * TOPK];
  __shared__ int il[TOPK];
  int tid = threadIdx.x;
  int row = blockIdx.x;               // b*S + t
  il[tid] = idxl[(size_t)row * TOPK + tid];
  {
    const u16* q = qb + (size_t)row * DMODEL;
    qs[tid] = bf2f(q[tid]);
    qs[tid + 512] = bf2f(q[tid + 512]);
  }
  __syncthreads();
  int h = tid >> 5;
  int lane = tid & 31;
  int b = row >> 11;                  // / S_LEN
  const u16* kbase = kb + (size_t)b * S_LEN * DMODEL + h * 64;
  const u16* vbase = vb + (size_t)b * S_LEN * DMODEL + h * 64;
  const float* qh = &qs[h * 64];
  float m = -1e30f;
  for (int jc = 0; jc < TOPK / 32; ++jc) {
    int j = jc * 32 + lane;
    int sidx = il[j];
    const u16* kr = kbase + (size_t)sidx * DMODEL;
    float acc = 0.f;
#pragma unroll
    for (int d8 = 0; d8 < 8; ++d8) {
      uint4 pk = *(const uint4*)(kr + d8 * 8);
      const float* qp = qh + d8 * 8;
      acc += qp[0] * bf2f(pk.x & 0xffffu);
      acc += qp[1] * bf2f(pk.x >> 16);
      acc += qp[2] * bf2f(pk.y & 0xffffu);
      acc += qp[3] * bf2f(pk.y >> 16);
      acc += qp[4] * bf2f(pk.z & 0xffffu);
      acc += qp[5] * bf2f(pk.z >> 16);
      acc += qp[6] * bf2f(pk.w & 0xffffu);
      acc += qp[7] * bf2f(pk.w >> 16);
    }
    float scv = acc * 0.125f;          // 1/sqrt(64)
    ps[h * TOPK + j] = scv;
    m = fmaxf(m, scv);
  }
#pragma unroll
  for (int off = 16; off; off >>= 1) m = fmaxf(m, __shfl_xor(m, off, 32));
  float lsum = 0.f;
  for (int jc = 0; jc < TOPK / 32; ++jc) {
    int j = jc * 32 + lane;
    float p = __expf(ps[h * TOPK + j] - m);
    ps[h * TOPK + j] = p;
    lsum += p;
  }
#pragma unroll
  for (int off = 16; off; off >>= 1) lsum += __shfl_xor(lsum, off, 32);
  float o0 = 0.f, o1 = 0.f;
  for (int j = 0; j < TOPK; ++j) {
    float p = ps[h * TOPK + j];        // wave-uniform LDS broadcast
    int sidx = il[j];
    const u16* vr = vbase + (size_t)sidx * DMODEL;
    o0 += p * bf2f((unsigned)vr[lane]);
    o1 += p * bf2f((unsigned)vr[lane + 32]);
  }
  float invl = 1.f / lsum;
  u16* outp = ob + (size_t)row * DMODEL + h * 64;
  outp[lane] = f2bf(o0 * invl);
  outp[lane + 32] = f2bf(o1 * invl);
}

// ---------------------------------------------------------------- host glue
extern "C" void kernel_launch(void* const* d_in, const int* in_sizes, int n_in,
                              void* d_out, int out_size, void* d_ws, size_t ws_size,
                              hipStream_t stream) {
  (void)in_sizes; (void)n_in; (void)out_size; (void)ws_size;
  const float* x     = (const float*)d_in[0];
  const float* w_qkv = (const float*)d_in[1];
  const float* w_o   = (const float*)d_in[2];
  const float* w_qi  = (const float*)d_in[3];
  const float* w_ki  = (const float*)d_in[4];
  const float* w_wi  = (const float*)d_in[5];

  char* base = (char*)d_ws;
  size_t off = 0;
  auto alloc = [&](size_t bytes) -> void* {
    void* p = base + off;
    off = (off + bytes + 255) & ~(size_t)255;
    return p;
  };
  u16* x_bf    = (u16*)alloc((size_t)ROWS * DMODEL * 2);
  u16* wqkv_bf = (u16*)alloc((size_t)3 * DMODEL * DMODEL * 2);
  u16* wo_bf   = (u16*)alloc((size_t)DMODEL * DMODEL * 2);
  u16* wqi_bf  = (u16*)alloc((size_t)HI_N * DI_N * DMODEL * 2);
  u16* wki_bf  = (u16*)alloc((size_t)DI_N * DMODEL * 2);
  float* qkv_f = (float*)alloc((size_t)ROWS * 3 * DMODEL * 4);   // reused for idx scores
  u16* q_bf    = (u16*)alloc((size_t)ROWS * DMODEL * 2);
  u16* k_bf    = (u16*)alloc((size_t)ROWS * DMODEL * 2);
  u16* v_bf    = (u16*)alloc((size_t)ROWS * DMODEL * 2);
  u16* qi_bf   = (u16*)alloc((size_t)ROWS * HI_N * DI_N * 2);
  u16* ki_bf   = (u16*)alloc((size_t)ROWS * DI_N * 2);
  float* wi_f  = (float*)alloc((size_t)ROWS * HI_N * 4);
  int* idxl    = (int*)alloc((size_t)ROWS * TOPK * 4);
  u16* attn_bf = (u16*)alloc((size_t)ROWS * DMODEL * 2);
  float* scores = qkv_f;   // alias: qkv fp32 is fully consumed by rope_split first

  auto cvt = [&](const float* in, u16* outp, size_t n) {
    cvt_f32_bf16<<<dim3((unsigned)((n + 255) / 256)), dim3(256), 0, stream>>>(in, outp, (int)n);
  };
  cvt(x, x_bf, (size_t)ROWS * DMODEL);
  cvt(w_qkv, wqkv_bf, (size_t)3 * DMODEL * DMODEL);
  cvt(w_o, wo_bf, (size_t)DMODEL * DMODEL);
  cvt(w_qi, wqi_bf, (size_t)HI_N * DI_N * DMODEL);
  cvt(w_ki, wki_bf, (size_t)DI_N * DMODEL);

  gemm_bf16_wmma<<<dim3(3 * DMODEL / 64, ROWS / 128), dim3(128), 0, stream>>>(
      x_bf, wqkv_bf, qkv_f, nullptr, ROWS, 3 * DMODEL, DMODEL, 0);
  gemm_bf16_wmma<<<dim3(HI_N * DI_N / 64, ROWS / 128), dim3(128), 0, stream>>>(
      x_bf, wqi_bf, nullptr, qi_bf, ROWS, HI_N * DI_N, DMODEL, 1);
  gemm_bf16_wmma<<<dim3(DI_N / 64, ROWS / 128), dim3(128), 0, stream>>>(
      x_bf, wki_bf, nullptr, ki_bf, ROWS, DI_N, DMODEL, 1);
  wi_proj<<<dim3(ROWS * HI_N / 256), dim3(256), 0, stream>>>(x, w_wi, wi_f);
  rope_split<<<dim3(ROWS * NHEADS * 32 / 256), dim3(256), 0, stream>>>(qkv_f, q_bf, k_bf, v_bf);
  indexer_scores<<<dim3(S_LEN / 16, S_LEN / 16, BATCH), dim3(32), 0, stream>>>(
      qi_bf, ki_bf, wi_f, scores);
  topk_select<<<dim3(ROWS), dim3(32), 0, stream>>>(scores, idxl);
  attention_sparse<<<dim3(ROWS), dim3(512), 0, stream>>>(q_bf, k_bf, v_bf, idxl, attn_bf);
  gemm_bf16_wmma<<<dim3(DMODEL / 64, ROWS / 128), dim3(128), 0, stream>>>(
      attn_bf, wo_bf, (float*)d_out, nullptr, ROWS, DMODEL, DMODEL, 0);
}
